// HandObjectCorrelationEncoder_66992899883222
// MI455X (gfx1250) — matrix operand order repaired and checked by reading, not compiled, not run on hardware
//
#include <hip/hip_runtime.h>
#include <hip/hip_bf16.h>
#include <stdint.h>

typedef __attribute__((ext_vector_type(16))) _Float16 v16h;
typedef __attribute__((ext_vector_type(8)))  _Float16 v8h;
typedef __attribute__((ext_vector_type(8)))  float    v8f;
typedef __attribute__((ext_vector_type(4)))  unsigned int u32x4;
typedef __attribute__((ext_vector_type(4)))  int i32x4;
typedef __attribute__((ext_vector_type(8)))  int i32x8;

#define FH 256
#define FW 256
#define IMG_H 512
#define IMG_W 512
#define NHAND 778
#define NOBJ 4096
#define NPTS (NHAND + NOBJ)
#define CFEAT 77
#define CPAD0 96
// LDS slab: 64 couts * cinp halves, TDM-padded 4 DWORDs every 16 DWORDs
// (each 32-half chunk occupies 40 halves). Max cinp=96 -> 192 chunks -> 7680.
#define SLAB_PAD_HALVES 7680

// ---------------------------------------------------------------- utilities
__global__ void zero_u32_kernel(uint32_t* __restrict__ p, long n) {
  long i = (long)blockIdx.x * blockDim.x + threadIdx.x;
  if (i < n) p[i] = 0u;
}

// ---------------------------------------------------- projection + bilinear
__global__ void sample_points_kernel(
    const float* __restrict__ fm, const float* __restrict__ R,
    const float* __restrict__ t, const float* __restrict__ Kc,
    const float* __restrict__ hand, const float* __restrict__ obj,
    const float* __restrict__ joints, const float* __restrict__ hlat,
    const float* __restrict__ olat, float* __restrict__ feats) {
  int pp = blockIdx.x;
  int tid = threadIdx.x;
  const float* psrc = (pp < NHAND) ? (hand + 3 * pp) : (obj + 3 * (pp - NHAND));
  float px = psrc[0], py = psrc[1], pz = psrc[2];
  if (tid < 64) {
    float acc = 0.f;
    for (int v = 0; v < 3; ++v) {
      const float* Rv = R + 9 * v;
      const float* tv = t + 3 * v;
      const float* Kv = Kc + 9 * v;
      float c0 = Rv[0] * px + Rv[1] * py + Rv[2] * pz + tv[0];
      float c1 = Rv[3] * px + Rv[4] * py + Rv[5] * pz + tv[1];
      float c2 = Rv[6] * px + Rv[7] * py + Rv[8] * pz + tv[2];
      float i0 = Kv[0] * c0 + Kv[1] * c1 + Kv[2] * c2;
      float i1 = Kv[3] * c0 + Kv[4] * c1 + Kv[5] * c2;
      float i2 = Kv[6] * c0 + Kv[7] * c1 + Kv[8] * c2;
      float u = i0 / i2, w = i1 / i2;
      float fx = u / (float)(IMG_W - 1) * (float)(FW - 1);
      float fy = w / (float)(IMG_H - 1) * (float)(FH - 1);
      float x0f = floorf(fx), y0f = floorf(fy);
      float wx = fx - x0f, wy = fy - y0f;
      int x0 = min(max((int)x0f, 0), FW - 1);
      int x1 = min(max((int)x0f + 1, 0), FW - 1);
      int y0 = min(max((int)y0f, 0), FH - 1);
      int y1 = min(max((int)y0f + 1, 0), FH - 1);
      const float* ch = fm + ((long)(v * 64 + tid)) * (FH * FW);
      float v00 = ch[y0 * FW + x0], v01 = ch[y0 * FW + x1];
      float v10 = ch[y1 * FW + x0], v11 = ch[y1 * FW + x1];
      acc += v00 * (1.f - wx) * (1.f - wy) + v01 * wx * (1.f - wy) +
             v10 * (1.f - wx) * wy + v11 * wx * wy;
    }
    feats[(long)pp * CFEAT + tid] = acc * (1.f / 3.f);
  } else if (tid < 67) {
    int d = tid - 64;
    float pv = (d == 0) ? px : ((d == 1) ? py : pz);
    feats[(long)pp * CFEAT + tid] = pv - joints[d];
  } else if (tid < 77) {
    int d = tid - 67;
    feats[(long)pp * CFEAT + tid] = (pp < NHAND) ? hlat[pp * 10 + d] : olat[d];
  }
}

// -------------------------------------------------------- scatter to volume
__global__ void scatter_kernel(const float* __restrict__ feats,
                               const int* __restrict__ hcoord,
                               const int* __restrict__ ocoord,
                               _Float16* __restrict__ vol,
                               float* __restrict__ mask0) {
  int pp = blockIdx.x, c = threadIdx.x;
  const int* crd = (pp < NHAND) ? (hcoord + 3 * pp) : (ocoord + 3 * (pp - NHAND));
  int vox = (crd[0] * 64 + crd[1]) * 64 + crd[2];
  if (c < CFEAT)
    vol[(long)vox * CPAD0 + c] = (_Float16)feats[(long)pp * CFEAT + c];
  else
    vol[(long)vox * CPAD0 + c] = (_Float16)0.f;
  if (c == 0) mask0[vox] = 1.f;
}

// ------------------------------- repack weights [tap][cout][cinp] + fold BN
__global__ void prep_weights_kernel(const float* __restrict__ w,
                                    const float* __restrict__ g,
                                    const float* __restrict__ b,
                                    const float* __restrict__ mu,
                                    const float* __restrict__ var,
                                    _Float16* __restrict__ wdst,
                                    float* __restrict__ bndst, int cin, int cinp) {
  int cout = blockIdx.x / 27;
  int tap = blockIdx.x % 27;
  int c = threadIdx.x;
  float val = (c < cin) ? w[((long)(cout * cin + c)) * 27 + tap] : 0.f;
  wdst[((long)(tap * 64 + cout)) * cinp + c] = (_Float16)val;
  if (tap == 0 && c == 0) {
    float s = g[cout] * rsqrtf(var[cout] + 1e-5f);
    bndst[cout] = s;
    bndst[64 + cout] = b[cout] - mu[cout] * s;
  }
}

// ------------------------------------------------------------- mask maxpool
__global__ void maxpool_mask_kernel(const float* __restrict__ m,
                                    float* __restrict__ mo, int So, int Si) {
  int i = blockIdx.x * blockDim.x + threadIdx.x;
  int tot = So * So * So;
  if (i >= tot) return;
  int x = i % So, y = (i / So) % So, z = i / (So * So);
  float best = 0.f;
  for (int dz = -1; dz <= 1; ++dz)
    for (int dy = -1; dy <= 1; ++dy)
      for (int dx = -1; dx <= 1; ++dx) {
        int xi = 2 * x + dx, yi = 2 * y + dy, zi = 2 * z + dz;
        if ((unsigned)xi < (unsigned)Si && (unsigned)yi < (unsigned)Si &&
            (unsigned)zi < (unsigned)Si)
          best = fmaxf(best, m[(zi * Si + yi) * Si + xi]);
      }
  mo[i] = best;
}

// --------------------------------------------------------------------------
// TDM: DMA one tap's weight slab (contiguous `halves` f16) into LDS.
// D# per CDNA5 ISA ch.8: group0 = {count/lds_addr/global_addr/type},
// group1 = {data_size=2B, pad every 16 DWORDs by 4 DWORDs, 1-D tile}.
// --------------------------------------------------------------------------
__device__ __forceinline__ void tdm_load_slab(const _Float16* gsrc,
                                              uint32_t lds_off, int halves) {
  uint64_t ga = (uint64_t)(uintptr_t)gsrc;
  u32x4 g0;
  g0[0] = 1u;                                    // count=1, user descriptor
  g0[1] = lds_off;                               // lds_addr (bytes)
  g0[2] = (uint32_t)ga;                          // global_addr[31:0]
  g0[3] = (uint32_t)((ga >> 32) & 0x01FFFFFFu)   // global_addr[56:32]
          | (2u << 30);                          // type = 2
  i32x8 g1;
  g1[0] = (int)((1u << 16)     // data_size = 2 bytes
              | (1u << 20)     // pad_enable (LDS bank de-conflict)
              | (3u << 22)     // pad_interval: every 16 DWORDs
              | (3u << 25));   // pad_amount: 4 DWORDs
  g1[1] = (int)(((uint32_t)halves & 0xFFFFu) << 16);                 // tensor_dim0 lo16
  g1[2] = (int)((((uint32_t)halves >> 16) & 0xFFFFu) | (1u << 16));  // dim0 hi | tensor_dim1=1
  g1[3] = (int)(((uint32_t)halves & 0xFFFFu) << 16);                 // tile_dim0
  g1[4] = 0;                                                         // tile_dim1/2 unused
  g1[5] = halves;                                                    // tensor_dim0_stride
  g1[6] = 0;
  g1[7] = 0;
  i32x4 gz = {0, 0, 0, 0};
#if defined(__clang_major__) && (__clang_major__ >= 23)
  i32x8 gz8 = {0, 0, 0, 0, 0, 0, 0, 0};
  __builtin_amdgcn_tensor_load_to_lds(g0, g1, gz, gz, gz8, 0);
#else
  __builtin_amdgcn_tensor_load_to_lds(g0, g1, gz, gz, 0);
#endif
}

// --------------------------------------------------------------------------
// Implicit-GEMM 3D conv (+BN+ReLU+mask) on WMMA, TDM-staged weights.
// in : [Sin^3][cinp] f16 ;  W : [27][64][cinp] f16 (unscaled, ch-padded; BN scale/shift in bn)
// out: [S^3][64] f16 ; fout (optional): channel-major f32 [64][S^3]
// Block = 256 thr = 8 waves; wave -> 16 voxels x all 64 couts (4 acc tiles):
// each B fragment (global, 2x16B) feeds 4 WMMAs; A fragments come from the
// double-buffered LDS slab (ds_load_b128) that wave 0 prefetches with the
// Tensor Data Mover one tap ahead (s_wait_tensorcnt + workgroup barrier).
// --------------------------------------------------------------------------
__global__ void __launch_bounds__(256)
conv3d_wmma_kernel(const _Float16* __restrict__ in, const _Float16* __restrict__ W,
                   const float* __restrict__ bn, const float* __restrict__ mask,
                   _Float16* __restrict__ out, float* __restrict__ fout,
                   int S, int Sin, int stride, int pad, int dil, int cinp) {
  __shared__ _Float16 smem[2 * SLAB_PAD_HALVES];
  const int tid = threadIdx.x;
  const int wave = tid >> 5;
  const int lane = tid & 31;
  const int hi = lane >> 4;
  const int nl = lane & 15;
  const int nvox = S * S * S;
  const int ntile = blockIdx.x * 8 + wave;
  const bool active = (ntile * 16) < nvox;       // wave-uniform
  const int n = min(ntile * 16 + nl, nvox - 1);  // output voxel (linear)
  const int x = n % S, y = (n / S) % S, z = n / (S * S);
  const int koffA = hi ? 8 : 0;
  const int koffB = hi ? 8 : 0;
  const int nchunks = cinp >> 5;
  const int slabHalves = cinp << 6;  // 64 * cinp (unpadded, global side)
  const uint32_t ldsBase = (uint32_t)(uintptr_t)(void*)smem;

  v8f acc[4] = {};
  const v8h vzero = {};

  if (wave == 0) {  // prefetch tap 0
    tdm_load_slab(W, ldsBase, slabHalves);
    __builtin_amdgcn_s_wait_tensorcnt(0);
  }
  __syncthreads();

  for (int tap = 0; tap < 27; ++tap) {
    if (wave == 0 && tap + 1 < 27)  // TDM prefetch next tap into other slab
      tdm_load_slab(W + (long)(tap + 1) * slabHalves,
                    ldsBase + (uint32_t)(((tap + 1) & 1) * (SLAB_PAD_HALVES * 2)),
                    slabHalves);
    if (active) {
      int tx = tap % 3, ty = (tap / 3) % 3, tz = tap / 9;
      int xi = x * stride - pad + tx * dil;
      int yi = y * stride - pad + ty * dil;
      int zi = z * stride - pad + tz * dil;
      bool inb = ((unsigned)xi < (unsigned)Sin) && ((unsigned)yi < (unsigned)Sin) &&
                 ((unsigned)zi < (unsigned)Sin);
      int xc = min(max(xi, 0), Sin - 1);
      int yc = min(max(yi, 0), Sin - 1);
      int zc = min(max(zi, 0), Sin - 1);
      const _Float16* bbase =
          in + ((long)((zc * Sin + yc) * Sin + xc)) * cinp + koffB;
      const _Float16* sA = smem + (size_t)(tap & 1) * SLAB_PAD_HALVES;
      __builtin_prefetch(bbase + cinp, 0, 1);  // -> global_prefetch_b8
      for (int ch = 0; ch < nchunks; ++ch) {
        v8h b0 = *(const v8h*)(bbase + ch * 32);
        v8h b1 = *(const v8h*)(bbase + ch * 32 + 16);
        b0 = inb ? b0 : vzero;  // halo zero-fill, EXEC stays all-ones
        b1 = inb ? b1 : vzero;
        union { v16h v; v8h h[2]; } B;
        B.h[0] = b0; B.h[1] = b1;
#pragma unroll
        for (int mt = 0; mt < 4; ++mt) {
          // chunk (mt*16+nl)*nchunks + ch occupies 40 halves in padded LDS
          const _Float16* ap =
              sA + (size_t)(((mt * 16 + nl) * nchunks + ch) * 40) + koffA;
          v8h a0 = *(const v8h*)ap;        // ds_load_b128
          v8h a1 = *(const v8h*)(ap + 16); // ds_load_b128
          union { v16h v; v8h h[2]; } A;
          A.h[0] = a0; A.h[1] = a1;
          acc[mt] = __builtin_amdgcn_wmma_f32_16x16x32_f16(
              false, A.v, false, B.v, (short)0, acc[mt], false, false);
        }
      }
    }
    if (wave == 0) __builtin_amdgcn_s_wait_tensorcnt(0);
    __syncthreads();
  }

  if (active) {
    const float mval = mask[n];
#pragma unroll
    for (int mt = 0; mt < 4; ++mt) {
      const int cbase = mt * 16 + hi * 8;
      v8h ov;
#pragma unroll
      for (int r = 0; r < 8; ++r) {
        float v = acc[mt][r] * bn[cbase + r] + bn[64 + cbase + r];
        v = fmaxf(v, 0.f) * mval;
        ov[r] = (_Float16)v;
        if (fout) fout[(long)(cbase + r) * nvox + n] = v;
      }
      *(v8h*)(out + (long)n * 64 + cbase) = ov;
    }
  }
}

// --------------------------------------------------------------------------
extern "C" void kernel_launch(void* const* d_in, const int* in_sizes, int n_in,
                              void* d_out, int out_size, void* d_ws, size_t ws_size,
                              hipStream_t stream) {
  (void)in_sizes; (void)out_size; (void)ws_size;
  const float* fm     = (const float*)d_in[0];
  const float* R      = (const float*)d_in[1];
  const float* t      = (const float*)d_in[2];
  const float* Kc     = (const float*)d_in[3];
  const float* hand   = (const float*)d_in[4];
  const float* joints = (const float*)d_in[5];
  const float* obj    = (const float*)d_in[6];
  const float* hlat   = (const float*)d_in[7];
  const float* olat   = (const float*)d_in[8];
  const int* hcoord   = (const int*)d_in[n_in - 2];
  const int* ocoord   = (const int*)d_in[n_in - 1];
  float* outp = (float*)d_out;

  char* wsp = (char*)d_ws;
  size_t off = 0;
  auto alloc = [&](size_t bytes) -> void* {
    off = (off + 255) & ~(size_t)255;
    void* p = wsp + off;
    off += bytes;
    return p;
  };

  _Float16* vol  = (_Float16*)alloc((size_t)262144 * CPAD0 * 2);
  _Float16* bufA = (_Float16*)alloc((size_t)262144 * 64 * 2);
  _Float16* bufB = (_Float16*)alloc((size_t)262144 * 64 * 2);
  float* feats = (float*)alloc((size_t)NPTS * CFEAT * 4);
  float* mask0 = (float*)alloc((size_t)262144 * 4);
  float* mask1 = (float*)alloc((size_t)32768 * 4);
  float* mask2 = (float*)alloc((size_t)4096 * 4);
  float* mask3 = (float*)alloc((size_t)512 * 4);
  float* mask4 = (float*)alloc((size_t)64 * 4);
  float* bnbuf = (float*)alloc((size_t)17 * 128 * 4);
  _Float16* wbuf[17];
  for (int i = 0; i < 17; ++i) {
    int cinp = (i == 0) ? CPAD0 : 64;
    wbuf[i] = (_Float16*)alloc((size_t)64 * 27 * cinp * 2);
  }

  // zero volume (covers pad channels + empty voxels) and mask0
  {
    long nu = (long)262144 * CPAD0 / 2;  // halves -> u32
    zero_u32_kernel<<<(int)((nu + 255) / 256), 256, 0, stream>>>((uint32_t*)vol, nu);
    zero_u32_kernel<<<(262144 + 255) / 256, 256, 0, stream>>>((uint32_t*)mask0, 262144);
  }

  // pack weights + fold BN (params leaves start at d_in[9], 5 per layer)
  for (int i = 0; i < 17; ++i) {
    int cin = (i == 0) ? CFEAT : 64;
    int cinp = (i == 0) ? CPAD0 : 64;
    prep_weights_kernel<<<64 * 27, cinp, 0, stream>>>(
        (const float*)d_in[9 + 5 * i + 0], (const float*)d_in[9 + 5 * i + 1],
        (const float*)d_in[9 + 5 * i + 2], (const float*)d_in[9 + 5 * i + 3],
        (const float*)d_in[9 + 5 * i + 4], wbuf[i], bnbuf + 128 * i, cin, cinp);
  }

  sample_points_kernel<<<NPTS, 128, 0, stream>>>(fm, R, t, Kc, hand, obj, joints,
                                                 hlat, olat, feats);
  scatter_kernel<<<NPTS, CPAD0, 0, stream>>>(feats, hcoord, ocoord, vol, mask0);

  auto conv = [&](int li, const _Float16* in, _Float16* ob, const float* mk,
                  int S, int Sin, int stride, int pad, int dil, int cinp, float* fo) {
    int blocks = (S * S * S + 127) / 128;
    conv3d_wmma_kernel<<<blocks, 256, 0, stream>>>(
        in, wbuf[li], bnbuf + 128 * li, mk, ob, fo, S, Sin, stride, pad, dil, cinp);
  };
  auto pool = [&](const float* mi, float* mo, int So, int Si) {
    int tot = So * So * So;
    maxpool_mask_kernel<<<(tot + 255) / 256, 256, 0, stream>>>(mi, mo, So, Si);
  };

  conv(0, vol,  bufA, mask0, 64, 64, 1, 1, 1, CPAD0, nullptr);     // conv0 a
  conv(1, bufA, bufB, mask0, 64, 64, 1, 1, 1, 64, nullptr);        // conv0 b
  pool(mask0, mask1, 32, 64);
  conv(2, bufB, bufA, mask1, 32, 64, 2, 1, 1, 64, nullptr);        // down0
  conv(3, bufA, bufB, mask1, 32, 32, 1, 1, 1, 64, nullptr);        // conv1 a
  conv(4, bufB, bufA, mask1, 32, 32, 1, 1, 1, 64, outp + 0);       // conv1 b -> out1
  pool(mask1, mask2, 16, 32);
  conv(5, bufA, bufB, mask2, 16, 32, 2, 1, 1, 64, nullptr);        // down1
  conv(6, bufB, bufA, mask2, 16, 16, 1, 2, 2, 64, nullptr);        // conv2 a
  conv(7, bufA, bufB, mask2, 16, 16, 1, 2, 2, 64, nullptr);        // conv2 b
  conv(8, bufB, bufA, mask2, 16, 16, 1, 2, 2, 64, outp + 2097152); // conv2 c -> out2
  pool(mask2, mask3, 8, 16);
  conv(9,  bufA, bufB, mask3, 8, 16, 2, 1, 1, 64, nullptr);        // down2
  conv(10, bufB, bufA, mask3, 8, 8, 1, 2, 2, 64, nullptr);         // conv3 a
  conv(11, bufA, bufB, mask3, 8, 8, 1, 2, 2, 64, nullptr);         // conv3 b
  conv(12, bufB, bufA, mask3, 8, 8, 1, 2, 2, 64, outp + 2359296);  // conv3 c -> out3
  pool(mask3, mask4, 4, 8);
  conv(13, bufA, bufB, mask4, 4, 8, 2, 1, 1, 64, nullptr);         // down3
  conv(14, bufB, bufA, mask4, 4, 4, 1, 2, 2, 64, nullptr);         // conv4 a
  conv(15, bufA, bufB, mask4, 4, 4, 1, 2, 2, 64, nullptr);         // conv4 b
  conv(16, bufB, bufA, mask4, 4, 4, 1, 2, 2, 64, outp + 2392064);  // conv4 c -> out4
}
